// FeedForwardECMoe_14628658610308
// MI455X (gfx1250) — matrix operand order repaired and checked by reading, not compiled, not run on hardware
//
#include <hip/hip_runtime.h>
#include <cstdint>
#include <cstddef>

using f16 = _Float16;
typedef __attribute__((ext_vector_type(16))) _Float16 v16h;
typedef __attribute__((ext_vector_type(8)))  _Float16 v8h;
typedef __attribute__((ext_vector_type(8)))  float    v8f;
typedef int v4i __attribute__((ext_vector_type(4)));

#define NB   8
#define T    1024
#define DIMD 1024
#define DIMF 4096
#define NE   8
#define KCAP 256
#define NEP  64
#define LDST 40   // padded LDS row stride in halves (80B: 16B aligned rows)

// ---------------- async global->LDS copy (GLOBAL_LOAD_ASYNC_TO_LDS_B128, ASYNCcnt) ----------------
__device__ inline void async_ld16(const f16* g, f16* l) {
  __builtin_amdgcn_global_load_async_to_lds_b128(
      (__attribute__((address_space(1))) v4i*)g,
      (__attribute__((address_space(3))) v4i*)l, 0, 0);
}
__device__ inline void wait_async0() {
  asm volatile("s_wait_asynccnt 0x0" ::: "memory");
}

// ---------------- zero output ----------------
__global__ void k_zero(float* __restrict__ p, int n4) {
  int i = (blockIdx.x * blockDim.x + threadIdx.x) * 4;
  if (i < n4) { p[i] = 0.f; p[i+1] = 0.f; p[i+2] = 0.f; p[i+3] = 0.f; }
}

// ---------------- gate + softmax ----------------
__global__ void k_gate_softmax(const float* __restrict__ x, const float* __restrict__ gw,
                               float* __restrict__ probs) {
  const int tok = blockIdx.x;              // 0 .. NB*T-1
  const int tid = threadIdx.x;             // 128
  const float* xr = x + (size_t)tok * DIMD;
  float acc[NE];
#pragma unroll
  for (int e = 0; e < NE; ++e) acc[e] = 0.f;
  for (int d0 = tid; d0 < DIMD; d0 += 128) {
    float xv = xr[d0];
#pragma unroll
    for (int e = 0; e < NE; ++e) acc[e] += xv * gw[e * DIMD + d0];
  }
  __shared__ float red[NE][128];
#pragma unroll
  for (int e = 0; e < NE; ++e) red[e][tid] = acc[e];
  __syncthreads();
  for (int s = 64; s > 0; s >>= 1) {
    if (tid < s) {
#pragma unroll
      for (int e = 0; e < NE; ++e) red[e][tid] += red[e][tid + s];
    }
    __syncthreads();
  }
  if (tid == 0) {
    float mx = red[0][0];
    for (int e = 1; e < NE; ++e) mx = fmaxf(mx, red[e][0]);
    float ex[NE]; float sum = 0.f;
    for (int e = 0; e < NE; ++e) { ex[e] = expf(red[e][0] - mx); sum += ex[e]; }
    float inv = 1.f / sum;
    for (int e = 0; e < NE; ++e) probs[(size_t)tok * NE + e] = ex[e] * inv;
  }
}

// ---------------- top-k via exact rank (ties -> lower index first, as jax top_k) ----------------
__global__ void k_topk(const float* __restrict__ probs, int* __restrict__ mIdx,
                       float* __restrict__ gVal) {
  const int z = blockIdx.x;                // n*8 + e
  const int n = z >> 3, e = z & 7;
  const int i = threadIdx.x;               // 0..1023
  __shared__ float p[T];
  p[i] = probs[((size_t)n * T + i) * NE + e];
  __syncthreads();
  const float pi = p[i];
  int rank = 0;
  for (int j = 0; j < T; ++j) {
    float pj = p[j];
    rank += (pj > pi) || (pj == pi && j < i);
  }
  if (rank < KCAP) {
    gVal[z * KCAP + rank] = pi;
    mIdx[z * KCAP + rank] = i;
  }
}

// ---------------- tiled transpose + f32->f16 convert: (B,R,C) -> (B,C,R) ----------------
__global__ void k_transpose_f32_f16(const float* __restrict__ in, f16* __restrict__ out,
                                    int R, int C) {
  __shared__ float tile[32][33];
  const size_t b = blockIdx.z;
  const float* ib = in + b * (size_t)R * C;
  f16* ob = out + b * (size_t)R * C;
  const int c  = blockIdx.x * 32 + threadIdx.x;
  const int r0 = blockIdx.y * 32;
#pragma unroll
  for (int j = 0; j < 4; ++j)
    tile[threadIdx.y + j * 8][threadIdx.x] = ib[(size_t)(r0 + threadIdx.y + j * 8) * C + c];
  __syncthreads();
  const int oc  = r0 + threadIdx.x;        // output inner index (= r)
  const int or0 = blockIdx.x * 32;         // output row base (= c)
#pragma unroll
  for (int j = 0; j < 4; ++j)
    ob[(size_t)(or0 + threadIdx.y + j * 8) * R + oc] = (f16)tile[threadIdx.x][threadIdx.y + j * 8];
}

// ---------------- gather selected tokens, convert to f16 ----------------
__global__ void k_gather(const float* __restrict__ x, const int* __restrict__ mIdx,
                         f16* __restrict__ xin) {
  const int idx = blockIdx.x;              // z*256 + slot
  const int z = idx >> 8, slot = idx & 255;
  const int n = z >> 3;
  const int tok = mIdx[z * KCAP + slot];
  const float* src = x + ((size_t)n * T + tok) * DIMD;
  f16* dst = xin + (size_t)idx * DIMD;
  for (int c = threadIdx.x; c < DIMD; c += 256) dst[c] = (f16)src[c];
}

// ---------------- WMMA fragment helpers (ISA 16-bit A/B VGPR layouts) ----------------
__device__ inline v16h frag16(const f16* p0, const f16* p1) {
  v8h lo = *(const v8h*)p0;
  v8h hi = *(const v8h*)p1;
  return __builtin_shufflevector(lo, hi, 0,1,2,3,4,5,6,7,8,9,10,11,12,13,14,15);
}

// fast GELU: x * sigmoid(1.5957691*(x + 0.044715 x^3)); |err| < ~1e-3, below f16 noise
__device__ inline float gelu_f(float v) {
  float u = 1.5957691216057308f * (v + 0.044715f * v * v * v);
  float s = 1.f / (1.f + __expf(-u));
  return v * s;
}

// ---------------- GEMM1: H[z] = gelu(xin[z] (256x1024) @ w1t[e] [N][K]) -> f16 ----------------
__global__ __launch_bounds__(256) void k_gemm1(const f16* __restrict__ xin,
                                               const f16* __restrict__ w1t,
                                               f16* __restrict__ H) {
  __shared__ __align__(16) f16 lsA[2][128 * LDST];
  __shared__ __align__(16) f16 lsB[2][128 * LDST];
  const int z = blockIdx.z;
  const int e = z & 7;
  const f16* A = xin + (size_t)z * KCAP * DIMD;     // 256 x 1024 [M][K]
  const f16* B = w1t + (size_t)e * DIMF * DIMD;     // 4096 x 1024 [N][K]
  const int t    = threadIdx.x;
  const int lane = t & 31;
  const int wid  = t >> 5;
  const int lm = lane & 15, hk = lane >> 4;
  const int m_off = (wid & 3) * 32, n_off = (wid >> 2) * 64;
  const int mtile = blockIdx.y * 128, ntile = blockIdx.x * 128;

  v8f acc[2][4] = {};

  const int r = t >> 1, cch = (t & 1) * 16;
  const f16* sa0 = A + (size_t)(mtile + r) * DIMD + cch;
  const f16* sb0 = B + (size_t)(ntile + r) * DIMD + cch;

  // prologue: async-stage K-tile 0 into buffer 0
  {
    async_ld16(sa0,      &lsA[0][r * LDST + cch]);
    async_ld16(sa0 + 8,  &lsA[0][r * LDST + cch + 8]);
    async_ld16(sb0,      &lsB[0][r * LDST + cch]);
    async_ld16(sb0 + 8,  &lsB[0][r * LDST + cch + 8]);
  }
  wait_async0();
  __syncthreads();

  int buf = 0;
  for (int kk = 0; kk < DIMD; kk += 32) {
    if (kk + 32 < DIMD) {           // async-prefetch next K-tile into other buffer
      int nb2 = buf ^ 1;
      async_ld16(sa0 + kk + 32,     &lsA[nb2][r * LDST + cch]);
      async_ld16(sa0 + kk + 40,     &lsA[nb2][r * LDST + cch + 8]);
      async_ld16(sb0 + kk + 32,     &lsB[nb2][r * LDST + cch]);
      async_ld16(sb0 + kk + 40,     &lsB[nb2][r * LDST + cch + 8]);
    }
    v16h afr[2], bfr[4];
#pragma unroll
    for (int i = 0; i < 2; ++i) {
      const f16* base = &lsA[buf][(m_off + i * 16 + lm) * LDST + hk * 8];
      afr[i] = frag16(base, base + 16);                 // K chunks [hk*8,+8) and [hk*8+16,+8)
    }
#pragma unroll
    for (int j = 0; j < 4; ++j) {
      const f16* base = &lsB[buf][(n_off + j * 16 + lm) * LDST + hk * 8];
      bfr[j] = frag16(base, base + 16);                 // K chunks [hk*8,+8) and [hk*8+16,+8)
    }
#pragma unroll
    for (int i = 0; i < 2; ++i)
#pragma unroll
      for (int j = 0; j < 4; ++j)
        acc[i][j] = __builtin_amdgcn_wmma_f32_16x16x32_f16(
            false, afr[i], false, bfr[j], (short)0, acc[i][j], false, false);
    wait_async0();
    __syncthreads();
    buf ^= 1;
  }

  f16* Hp = H + (size_t)z * KCAP * DIMF;
#pragma unroll
  for (int i = 0; i < 2; ++i) {
#pragma unroll
    for (int j = 0; j < 4; ++j) {
      const int col   = ntile + n_off + j * 16 + lm;
      const int rbase = mtile + m_off + i * 16 + hk * 8;
#pragma unroll
      for (int vr = 0; vr < 8; ++vr) {
        float v = acc[i][j][vr];
        Hp[(size_t)(rbase + vr) * DIMF + col] = (f16)gelu_f(v);
      }
    }
  }
}

// ---------------- GEMM2: out += g * (H[z] (256x4096) @ w2t[e] [N=d][K=f]) scatter-add ----------------
__global__ __launch_bounds__(256) void k_gemm2(const f16* __restrict__ H,
                                               const f16* __restrict__ w2t,
                                               const int* __restrict__ mIdx,
                                               const float* __restrict__ gVal,
                                               float* __restrict__ out) {
  __shared__ __align__(16) f16 lsA[2][128 * LDST];
  __shared__ __align__(16) f16 lsB[2][128 * LDST];
  __shared__ int   tok_s[128];
  __shared__ float gv_s[128];
  const int z = blockIdx.z;
  const int e = z & 7, nb = z >> 3;
  const f16* A = H   + (size_t)z * KCAP * DIMF;     // 256 x 4096 [M][K]
  const f16* B = w2t + (size_t)e * DIMD * DIMF;     // 1024 x 4096 [N][K]
  const int t    = threadIdx.x;
  const int lane = t & 31;
  const int wid  = t >> 5;
  const int lm = lane & 15, hk = lane >> 4;
  const int m_off = (wid & 3) * 32, n_off = (wid >> 2) * 64;
  const int mtile = blockIdx.y * 128, ntile = blockIdx.x * 128;

  if (t < 128) {
    tok_s[t] = mIdx[z * KCAP + mtile + t];
    gv_s[t]  = gVal[z * KCAP + mtile + t];
  }

  v8f acc[2][4] = {};

  const int r = t >> 1, cch = (t & 1) * 16;
  const f16* sa0 = A + (size_t)(mtile + r) * DIMF + cch;
  const f16* sb0 = B + (size_t)(ntile + r) * DIMF + cch;

  {
    async_ld16(sa0,      &lsA[0][r * LDST + cch]);
    async_ld16(sa0 + 8,  &lsA[0][r * LDST + cch + 8]);
    async_ld16(sb0,      &lsB[0][r * LDST + cch]);
    async_ld16(sb0 + 8,  &lsB[0][r * LDST + cch + 8]);
  }
  wait_async0();
  __syncthreads();

  int buf = 0;
  for (int kk = 0; kk < DIMF; kk += 32) {
    if (kk + 32 < DIMF) {
      int nb2 = buf ^ 1;
      async_ld16(sa0 + kk + 32,     &lsA[nb2][r * LDST + cch]);
      async_ld16(sa0 + kk + 40,     &lsA[nb2][r * LDST + cch + 8]);
      async_ld16(sb0 + kk + 32,     &lsB[nb2][r * LDST + cch]);
      async_ld16(sb0 + kk + 40,     &lsB[nb2][r * LDST + cch + 8]);
    }
    v16h afr[2], bfr[4];
#pragma unroll
    for (int i = 0; i < 2; ++i) {
      const f16* base = &lsA[buf][(m_off + i * 16 + lm) * LDST + hk * 8];
      afr[i] = frag16(base, base + 16);
    }
#pragma unroll
    for (int j = 0; j < 4; ++j) {
      const f16* base = &lsB[buf][(n_off + j * 16 + lm) * LDST + hk * 8];
      bfr[j] = frag16(base, base + 16);
    }
#pragma unroll
    for (int i = 0; i < 2; ++i)
#pragma unroll
      for (int j = 0; j < 4; ++j)
        acc[i][j] = __builtin_amdgcn_wmma_f32_16x16x32_f16(
            false, afr[i], false, bfr[j], (short)0, acc[i][j], false, false);
    wait_async0();
    __syncthreads();
    buf ^= 1;
  }

  float* outp = out + (size_t)nb * T * DIMD;
#pragma unroll
  for (int i = 0; i < 2; ++i) {
#pragma unroll
    for (int j = 0; j < 4; ++j) {
      const int col  = ntile + n_off + j * 16 + lm;
      const int rloc = m_off + i * 16 + hk * 8;     // row within this 128-row tile
#pragma unroll
      for (int vr = 0; vr < 8; ++vr) {
        const int rl = rloc + vr;
        const int token = tok_s[rl];
        float v = acc[i][j][vr] * gv_s[rl];
        unsafeAtomicAdd(outp + (size_t)token * DIMD + col, v);   // global_atomic_add_f32
      }
    }
  }
}

// ---------------- launch ----------------
extern "C" void kernel_launch(void* const* d_in, const int* in_sizes, int n_in,
                              void* d_out, int out_size, void* d_ws, size_t ws_size,
                              hipStream_t stream) {
  (void)in_sizes; (void)n_in; (void)out_size; (void)ws_size;
  const float* x  = (const float*)d_in[0];
  const float* gw = (const float*)d_in[1];
  const float* w1 = (const float*)d_in[2];
  const float* w2 = (const float*)d_in[3];
  float* out = (float*)d_out;

  char* ws = (char*)d_ws;
  float* probs = (float*)(ws + 0);             // 256 KB
  int*   mIdx  = (int*)  (ws + 262144);        // 64 KB
  float* gVal  = (float*)(ws + 327680);        // 64 KB
  f16*   w1t   = (f16*)  (ws + 393216);        // 64 MB  (e, f, d)
  f16*   w2t   = (f16*)  (ws + 67502080);      // 64 MB  (e, d, f)
  f16*   xin   = (f16*)  (ws + 134610944);     // 32 MB  (z, 256, d)
  f16*   Hbuf  = (f16*)  (ws + 168165376);     // 128 MB (z, 256, f)

  k_zero<<<(NB * T * DIMD) / 4 / 256, 256, 0, stream>>>(out, NB * T * DIMD);
  k_gate_softmax<<<NB * T, 128, 0, stream>>>(x, gw, probs);
  k_topk<<<NEP, 1024, 0, stream>>>(probs, mIdx, gVal);
  k_transpose_f32_f16<<<dim3(DIMF / 32, DIMD / 32, NE), dim3(32, 8), 0, stream>>>(w1, w1t, DIMD, DIMF);
  k_transpose_f32_f16<<<dim3(DIMD / 32, DIMF / 32, NE), dim3(32, 8), 0, stream>>>(w2, w2t, DIMF, DIMD);
  k_gather<<<NEP * KCAP, 256, 0, stream>>>(x, mIdx, xin);
  k_gemm1<<<dim3(DIMF / 128, KCAP / 128, NEP), 256, 0, stream>>>(xin, w1t, Hbuf);
  k_gemm2<<<dim3(DIMD / 128, KCAP / 128, NEP), 256, 0, stream>>>(Hbuf, w2t, mIdx, gVal, out);
}
